// MultiHeadAttention_17093969838338
// MI455X (gfx1250) — matrix operand order repaired and checked by reading, not compiled, not run on hardware
//
#include <hip/hip_runtime.h>
#include <hip/hip_bf16.h>

typedef __attribute__((ext_vector_type(16))) _Float16 v16h;
typedef __attribute__((ext_vector_type(8)))  float    v8f;

#define B_   2
#define S_   2048
#define D_   1024
#define H_   16
#define DK_  64
#define EPS_ 1e-5f

#define SSTR 2052                       // padded LDS row stride (floats): breaks bank conflicts
#define SMEM_FLOATS (16 * SSTR + 8 * 16 * 64)

// ---------------------------------------------------------------------------
// WMMA fragment loaders, wave32; A/B use one contiguous k order, C/D the ISA layout.
// A 16x32 f16: lane L holds row m=L&15; element e holds
//   k = e + (L&16)
// B 32x16 f16: lane L holds col n=L&15; element e holds k = e + (L&16)
// C/D 16x16 f32: lane L col n=L&15; vreg r holds row m = r + ((L&16)>>1)
// ---------------------------------------------------------------------------

static __device__ __forceinline__ v16h a_frag_f32(const float* p0, int ld, int lane) {
  const float* p = p0 + (size_t)(lane & 15) * ld + (lane & 16);
  v16h a;
#pragma unroll
  for (int e = 0; e < 16; ++e) a[e] = (_Float16)p[e];
  return a;
}

static __device__ __forceinline__ v16h a_frag_f16(const _Float16* p0, int ld, int lane) {
  const _Float16* p = p0 + (size_t)(lane & 15) * ld + (lane & 16);
  v16h a;
#pragma unroll
  for (int e = 0; e < 16; ++e) a[e] = p[e];
  return a;
}

// B = M^T where M is row-major: B[k][n] = M[n][k]; per-lane contiguous in k.
static __device__ __forceinline__ v16h bT_frag_f32(const float* p0, int ld, int lane) {
  const float* p = p0 + (size_t)(lane & 15) * ld + (lane & 16);
  v16h b;
#pragma unroll
  for (int e = 0; e < 16; ++e) b[e] = (_Float16)p[e];
  return b;
}

static __device__ __forceinline__ v16h bT_frag_f16(const _Float16* p0, int ld, int lane) {
  const _Float16* p = p0 + (size_t)(lane & 15) * ld + (lane & 16);
  v16h b;
#pragma unroll
  for (int e = 0; e < 16; ++e) b[e] = p[e];
  return b;
}

static __device__ __forceinline__ v8f wmma_f16(v16h a, v16h b, v8f c) {
  return __builtin_amdgcn_wmma_f32_16x16x32_f16(false, a, false, b, (short)0, c,
                                                false, false);
}

// ---------------------------------------------------------------------------
// Kernel 1: X @ W^T + bias -> f16, head-split [B,H,S,DK] or transposed
// [B,H,DK,S] (for V, so the ctx-phase B fragments are per-lane contiguous).
// one wave per 16x64 output strip; K loop = 32 steps of 32
// ---------------------------------------------------------------------------
__global__ __launch_bounds__(32) void qkv_kernel(const float* __restrict__ X,
                                                 const float* __restrict__ W,
                                                 const float* __restrict__ bias,
                                                 _Float16* __restrict__ out,
                                                 int transposeOut) {
  const int lane = threadIdx.x;
  const int mt   = blockIdx.x;       // row tile over B*S (256 tiles)
  const int n0   = blockIdx.y * 64;  // output-feature strip

  const float* Abase = X + (size_t)mt * 16 * D_;
  v8f acc[4] = {};

  for (int kt = 0; kt < D_; kt += 32) {
    __builtin_prefetch(Abase + kt + 128, 0, 0);   // global_prefetch_b8
    v16h a = a_frag_f32(Abase + kt, D_, lane);
#pragma unroll
    for (int j = 0; j < 4; ++j) {
      v16h b = bT_frag_f32(W + (size_t)(n0 + 16 * j) * D_ + kt, D_, lane);
      acc[j] = wmma_f16(a, b, acc[j]);
    }
  }

#pragma unroll
  for (int j = 0; j < 4; ++j) {
    const int col = n0 + 16 * j + (lane & 15);
    const int h = col >> 6, dk = col & 63;
    const float bb = bias[col];
#pragma unroll
    for (int r = 0; r < 8; ++r) {
      const int grow = mt * 16 + r + ((lane & 16) >> 1);   // global row in B*S
      const int bi = grow >> 11;                            // / S_
      const int s  = grow & (S_ - 1);
      const _Float16 val = (_Float16)(acc[j][r] + bb);
      if (transposeOut)
        out[((((size_t)bi * H_ + h) * DK_ + dk) * S_) + s] = val;   // [B,H,DK,S]
      else
        out[((((size_t)bi * H_ + h) * S_ + s) * DK_) + dk] = val;   // [B,H,S,DK]
    }
  }
}

// ---------------------------------------------------------------------------
// Kernel 2 (fused): per (q-tile, head) workgroup of 8 waves.
//   phase 1: scores = (Qh @ Kh^T)*scale masked  -> LDS strip [16][2048]
//   phase 2: row softmax in LDS (shfl reductions), single prob write to d_out
//   phase 3: ctx = P @ V with WMMA A-fragments read from LDS and B fragments
//            from the pre-transposed V^T (contiguous per-lane b128 loads);
//            8 per-wave K-partials reduced through LDS
// Attention matrix touches HBM exactly once (the mandatory output write).
// ---------------------------------------------------------------------------
__global__ __launch_bounds__(256) void attn_fused_kernel(const _Float16* __restrict__ Qh,
                                                         const _Float16* __restrict__ Kh,
                                                         const _Float16* __restrict__ VhT,
                                                         const int* __restrict__ mask,
                                                         float* __restrict__ attn,
                                                         float* __restrict__ ctx) {
  extern __shared__ float smem[];
  float* sS = smem;                 // [16][SSTR] score/prob strip
  float* sP = smem + 16 * SSTR;     // [8][16][64] ctx partials

  const int t    = threadIdx.x;
  const int lane = t & 31;
  const int w    = t >> 5;          // wave id 0..7
  const int qt   = blockIdx.x;      // 0..127
  const int bh   = blockIdx.y;      // 0..31
  const int b    = bh >> 4, h = bh & 15;

  const _Float16* Q  = Qh  + (size_t)bh * S_ * DK_ + (size_t)qt * 16 * DK_;
  const _Float16* K  = Kh  + (size_t)bh * S_ * DK_;
  const _Float16* VT = VhT + (size_t)bh * DK_ * S_;   // [DK][S]

  // ---- phase 1: scores for keys [w*256, (w+1)*256) --------------------------
  const v16h a0 = a_frag_f16(Q + 0,  DK_, lane);
  const v16h a1 = a_frag_f16(Q + 32, DK_, lane);

  const float scale = 0.125f;  // 1/sqrt(64)
#pragma unroll
  for (int nt = 0; nt < 16; ++nt) {
    const int n0 = w * 256 + nt * 16;
    v8f acc = {};
    acc = wmma_f16(a0, bT_frag_f16(K + (size_t)n0 * DK_ + 0,  DK_, lane), acc);
    acc = wmma_f16(a1, bT_frag_f16(K + (size_t)n0 * DK_ + 32, DK_, lane), acc);

    const int col = n0 + (lane & 15);
#pragma unroll
    for (int r = 0; r < 8; ++r) {
      const int m  = r + ((lane & 16) >> 1);
      const int qg = qt * 16 + m;
      const int mv = mask[((size_t)b * S_ + qg) * S_ + col];
      sS[m * SSTR + col] = (mv == 0) ? -1e9f : acc[r] * scale;
    }
  }
  __syncthreads();

  // ---- phase 2: softmax of row r (threads r*16..r*16+15 cooperate) ----------
  {
    const int r = t >> 4, cid = t & 15;
    float* rowS = sS + r * SSTR;
    float* gRow = attn + (size_t)bh * S_ * S_ + (size_t)(qt * 16 + r) * S_;

    float mx = -3.4e38f;
#pragma unroll 8
    for (int i = 0; i < 128; ++i) mx = fmaxf(mx, rowS[cid + 16 * i]);
#pragma unroll
    for (int off = 1; off < 16; off <<= 1) mx = fmaxf(mx, __shfl_xor(mx, off, 32));

    float sum = 0.f;
#pragma unroll 8
    for (int i = 0; i < 128; ++i) {
      const float e = __expf(rowS[cid + 16 * i] - mx);
      rowS[cid + 16 * i] = e;
      sum += e;
    }
#pragma unroll
    for (int off = 1; off < 16; off <<= 1) sum += __shfl_xor(sum, off, 32);
    const float inv = 1.0f / sum;

#pragma unroll 8
    for (int i = 0; i < 128; ++i) {
      const float p = rowS[cid + 16 * i] * inv;
      rowS[cid + 16 * i] = p;
      gRow[cid + 16 * i] = p;      // the single HBM write of the prob matrix
    }
  }
  __syncthreads();

  // ---- phase 3: ctx partials, wave w covers keys [w*256, (w+1)*256) ---------
  {
    v8f acc[4] = {};
    for (int ks = 0; ks < 256; ks += 32) {
      const int kg = w * 256 + ks;
      v16h a = a_frag_f32(sS + kg, SSTR, lane);   // probs from LDS -> f16
#pragma unroll
      for (int j = 0; j < 4; ++j) {
        // B[k][d] = VT[d][k]: per-lane contiguous 32B from the transposed V
        v16h bf = bT_frag_f16(VT + (size_t)(16 * j) * S_ + kg, S_, lane);
        acc[j] = wmma_f16(a, bf, acc[j]);
      }
    }
#pragma unroll
    for (int j = 0; j < 4; ++j) {
      const int d = 16 * j + (lane & 15);
#pragma unroll
      for (int r = 0; r < 8; ++r) {
        const int m = r + ((lane & 16) >> 1);
        sP[((w * 16 + m) << 6) + d] = acc[j][r];
      }
    }
  }
  __syncthreads();

  // ---- reduce 8 wave-partials, write merged-head ctx ------------------------
  for (int o = t; o < 16 * 64; o += 256) {
    const int m = o >> 6, d = o & 63;
    float s = 0.f;
#pragma unroll
    for (int ww = 0; ww < 8; ++ww) s += sP[((ww * 16 + m) << 6) + d];
    ctx[(((size_t)b * S_ + qt * 16 + m) * D_) + h * DK_ + d] = s;
  }
}

// ---------------------------------------------------------------------------
// Kernel 3: y = ctx @ Wo^T + bo + residual
// ---------------------------------------------------------------------------
__global__ __launch_bounds__(32) void oproj_kernel(const float* __restrict__ ctx,
                                                   const float* __restrict__ Wo,
                                                   const float* __restrict__ bo,
                                                   const float* __restrict__ resid,
                                                   float* __restrict__ y) {
  const int lane = threadIdx.x;
  const int mt = blockIdx.x;
  const int n0 = blockIdx.y * 64;

  const float* Abase = ctx + (size_t)mt * 16 * D_;
  v8f acc[4] = {};

  for (int kt = 0; kt < D_; kt += 32) {
    __builtin_prefetch(Abase + kt + 128, 0, 0);
    v16h a = a_frag_f32(Abase + kt, D_, lane);
#pragma unroll
    for (int j = 0; j < 4; ++j) {
      v16h bf = bT_frag_f32(Wo + (size_t)(n0 + 16 * j) * D_ + kt, D_, lane);
      acc[j] = wmma_f16(a, bf, acc[j]);
    }
  }

#pragma unroll
  for (int j = 0; j < 4; ++j) {
    const int col = n0 + 16 * j + (lane & 15);
    const float bb = bo[col];
#pragma unroll
    for (int r = 0; r < 8; ++r) {
      const int grow = mt * 16 + r + ((lane & 16) >> 1);
      const size_t o = (size_t)grow * D_ + col;
      y[o] = acc[j][r] + bb + resid[o];
    }
  }
}

// ---------------------------------------------------------------------------
// Kernel 4: LayerNorm per row of D=1024
// ---------------------------------------------------------------------------
__global__ __launch_bounds__(256) void ln_kernel(const float* __restrict__ y,
                                                 const float* __restrict__ gamma,
                                                 const float* __restrict__ beta,
                                                 float* __restrict__ out) {
  __shared__ float red[256];
  const size_t row = blockIdx.x;
  const float* p = y + row * D_;
  const int t = threadIdx.x;

  float v[4];
  float s = 0.f;
#pragma unroll
  for (int i = 0; i < 4; ++i) { v[i] = p[t + i * 256]; s += v[i]; }
  red[t] = s; __syncthreads();
  for (int w = 128; w > 0; w >>= 1) { if (t < w) red[t] += red[t + w]; __syncthreads(); }
  const float mu = red[0] * (1.0f / D_); __syncthreads();

  float ss = 0.f;
#pragma unroll
  for (int i = 0; i < 4; ++i) { const float d = v[i] - mu; ss += d * d; }
  red[t] = ss; __syncthreads();
  for (int w = 128; w > 0; w >>= 1) { if (t < w) red[t] += red[t + w]; __syncthreads(); }
  const float inv = rsqrtf(red[0] * (1.0f / D_) + EPS_);

  float* o = out + row * D_;
#pragma unroll
  for (int i = 0; i < 4; ++i) {
    const int c = t + i * 256;
    o[c] = (v[i] - mu) * inv * gamma[c] + beta[c];
  }
}

// ---------------------------------------------------------------------------
extern "C" void kernel_launch(void* const* d_in, const int* in_sizes, int n_in,
                              void* d_out, int out_size, void* d_ws, size_t ws_size,
                              hipStream_t stream) {
  (void)in_sizes; (void)n_in; (void)out_size; (void)ws_size;

  const float* q    = (const float*)d_in[0];
  const float* k    = (const float*)d_in[1];
  const float* v    = (const float*)d_in[2];
  const int*   mask = (const int*)  d_in[3];
  const float* Wq   = (const float*)d_in[4];
  const float* bq   = (const float*)d_in[5];
  const float* Wk   = (const float*)d_in[6];
  const float* bk   = (const float*)d_in[7];
  const float* Wv   = (const float*)d_in[8];
  const float* bv   = (const float*)d_in[9];
  const float* Wo   = (const float*)d_in[10];
  const float* bo   = (const float*)d_in[11];
  const float* gamma= (const float*)d_in[12];
  const float* beta = (const float*)d_in[13];

  float* out  = (float*)d_out;                       // [B,S,D]
  float* attn = out + (size_t)B_ * S_ * D_;          // [B,H,S,S]

  const size_t headElems = (size_t)B_ * H_ * S_ * DK_;  // 4M
  _Float16* Qh  = (_Float16*)d_ws;
  _Float16* Kh  = Qh + headElems;
  _Float16* VhT = Kh + headElems;                        // [B,H,DK,S]
  float* ctx = (float*)((char*)d_ws + 3 * headElems * sizeof(_Float16));  // +24MB
  float* y   = (float*)d_ws;  // reuse Qh/Kh region (dead after attn_fused)

  const dim3 gGemm(256, 16);   // (B*S)/16 row tiles x D/64 col strips
  const dim3 bWave(32);

  qkv_kernel<<<gGemm, bWave, 0, stream>>>(q, Wq, bq, Qh, 0);
  qkv_kernel<<<gGemm, bWave, 0, stream>>>(k, Wk, bk, Kh, 0);
  qkv_kernel<<<gGemm, bWave, 0, stream>>>(v, Wv, bv, VhT, 1);

  attn_fused_kernel<<<dim3(S_ / 16, B_ * H_), dim3(256),
                      SMEM_FLOATS * sizeof(float), stream>>>(Qh, Kh, VhT, mask, attn, ctx);

  oproj_kernel<<<gGemm, bWave, 0, stream>>>(ctx, Wo, bo, q, y);
  ln_kernel<<<dim3(B_ * S_), dim3(256), 0, stream>>>(y, gamma, beta, out);
}
